// MultiHeadAttention_6468220748710
// MI455X (gfx1250) — compile-verified, matrix operand order checked by reading, not run on hardware
//
#include <hip/hip_runtime.h>
#include <hip/hip_bf16.h>

// ---- problem constants (match reference) ----
#define Bc   2
#define Sc   2048
#define Dm   1024
#define Hc   16
#define HDc  64
#define BSM  (Bc * Sc)          // 4096 rows

typedef __attribute__((ext_vector_type(16))) _Float16 v16h;
typedef __attribute__((ext_vector_type(8)))  _Float16 v8h;
typedef __attribute__((ext_vector_type(4)))  _Float16 v4h;
typedef __attribute__((ext_vector_type(8)))  float    v8f;
typedef __attribute__((ext_vector_type(4)))  float    v4f;
typedef __attribute__((ext_vector_type(4)))  int      v4i;

// ---------- CDNA5 async global->LDS copy (ASYNCcnt path), guarded ----------
#if defined(__has_builtin)
#  if __has_builtin(__builtin_amdgcn_global_load_async_to_lds_b128)
#    define HAS_ASYNC_CP 1
#  endif
#endif
#ifndef HAS_ASYNC_CP
#  define HAS_ASYNC_CP 0
#endif

#if HAS_ASYNC_CP
typedef __attribute__((address_space(1))) v4i* g_v4i_p;   // global int4*
typedef __attribute__((address_space(3))) v4i* l_v4i_p;   // LDS int4*
__device__ __forceinline__ void cp_b128(void* lds, const void* g) {
    // global side: flat address == global address, so inttoptr into AS1 is safe;
    // LDS side: generic -> AS3 addrspacecast.
    __builtin_amdgcn_global_load_async_to_lds_b128(
        (g_v4i_p)(uintptr_t)g, (l_v4i_p)lds, 0, 0);
}
__device__ __forceinline__ void cp_wait() {
#  if __has_builtin(__builtin_amdgcn_s_wait_asynccnt)
    __builtin_amdgcn_s_wait_asynccnt(0);
#  else
    asm volatile("s_wait_asynccnt 0" ::: "memory");
#  endif
}
#else
__device__ __forceinline__ void cp_b128(void* lds, const void* g) {
    *(v8h*)lds = *(const v8h*)g;
}
__device__ __forceinline__ void cp_wait() {}
#endif

__device__ __forceinline__ v16h make_frag(const _Float16* p) {
    // per-lane 16-bit fragment = two contiguous 8-half (16B) chunks:
    // k in [8*hi, 8*hi+8) and [16+8*hi, 16+8*hi+8); caller bakes 8*hi into p.
    v8h lo = *(const v8h*)(p);
    v8h hi8 = *(const v8h*)(p + 16);
    return __builtin_shufflevector(lo, hi8, 0,1,2,3,4,5,6,7,8,9,10,11,12,13,14,15);
}

#define GM   128   // block M tile (2 x 16 rows per wave)
#define GN   64
#define GK   32

// ---------------- GEMM: C_f16 = (A_f32 * W_f32 + bias) * scale ----------------
__global__ __launch_bounds__(128) void gemm_proj_f16out(
    const float* __restrict__ A, const float* __restrict__ W,
    const float* __restrict__ bias, _Float16* __restrict__ Cout,
    float scale, int M, int N, int Kdim)
{
    __shared__ __align__(16) _Float16 ldsA[GM * GK];   // [m][k], stride 32
    __shared__ __align__(16) _Float16 ldsBt[GN * GK];  // [n][k], stride 32

    const int tid  = threadIdx.x;
    const int wave = tid >> 5;
    const int lane = tid & 31;
    const int r    = lane & 15;
    const int hi   = lane >> 4;

    const int m0 = blockIdx.y * GM;
    const int n0 = blockIdx.x * GN;

    v8f acc[2][4] = {};

    for (int k0 = 0; k0 < Kdim; k0 += GK) {
        // A tile: 128x32 f32 -> f16 (1024 v4f, 8 per thread)
        #pragma unroll
        for (int j = 0; j < 8; ++j) {
            int idx = tid + j * 128;
            int row = idx >> 3;
            int c4  = (idx & 7) << 2;
            v4f a4 = *(const v4f*)(A + (size_t)(m0 + row) * Kdim + k0 + c4);
            v4h h4;
            h4[0] = (_Float16)a4[0]; h4[1] = (_Float16)a4[1];
            h4[2] = (_Float16)a4[2]; h4[3] = (_Float16)a4[3];
            *(v4h*)(&ldsA[row * GK + c4]) = h4;
        }
        // W tile: 32x64 f32, transposed into [n][k]
        #pragma unroll
        for (int j = 0; j < 4; ++j) {
            int idx  = tid + j * 128;
            int krow = idx >> 4;
            int c4   = (idx & 15) << 2;
            v4f b4 = *(const v4f*)(W + (size_t)(k0 + krow) * N + n0 + c4);
            ldsBt[(c4 + 0) * GK + krow] = (_Float16)b4[0];
            ldsBt[(c4 + 1) * GK + krow] = (_Float16)b4[1];
            ldsBt[(c4 + 2) * GK + krow] = (_Float16)b4[2];
            ldsBt[(c4 + 3) * GK + krow] = (_Float16)b4[3];
        }
        __syncthreads();

        v16h af[2];
        #pragma unroll
        for (int mi = 0; mi < 2; ++mi)
            af[mi] = make_frag(&ldsA[(wave * 32 + mi * 16 + r) * GK + 8 * hi]);
        #pragma unroll
        for (int sub = 0; sub < 4; ++sub) {
            v16h bfrag = make_frag(&ldsBt[(sub * 16 + r) * GK + 8 * hi]);
            #pragma unroll
            for (int mi = 0; mi < 2; ++mi)
                acc[mi][sub] = __builtin_amdgcn_wmma_f32_16x16x32_f16(
                    false, af[mi], false, bfrag, (short)0, acc[mi][sub], false, false);
        }
        __syncthreads();
    }

    #pragma unroll
    for (int mi = 0; mi < 2; ++mi) {
        #pragma unroll
        for (int sub = 0; sub < 4; ++sub) {
            int col = n0 + sub * 16 + r;
            float bv = bias[col];
            #pragma unroll
            for (int v = 0; v < 8; ++v) {
                int row = m0 + wave * 32 + mi * 16 + hi * 8 + v;
                Cout[(size_t)row * N + col] = (_Float16)((acc[mi][sub][v] + bv) * scale);
            }
        }
    }
}

// ---------------- GEMM: out_f32 = A_f16 * W_f32 + bias ----------------
__global__ __launch_bounds__(128) void gemm_out_f32(
    const _Float16* __restrict__ A, const float* __restrict__ W,
    const float* __restrict__ bias, float* __restrict__ Out,
    int M, int N, int Kdim)
{
    __shared__ __align__(16) _Float16 ldsA[GM * GK];
    __shared__ __align__(16) _Float16 ldsBt[GN * GK];

    const int tid  = threadIdx.x;
    const int wave = tid >> 5;
    const int lane = tid & 31;
    const int r    = lane & 15;
    const int hi   = lane >> 4;

    const int m0 = blockIdx.y * GM;
    const int n0 = blockIdx.x * GN;

    v8f acc[2][4] = {};

    for (int k0 = 0; k0 < Kdim; k0 += GK) {
        // A tile: 128x32 f16 straight copy -> async to LDS (512 v8h, 4/thread)
        #pragma unroll
        for (int j = 0; j < 4; ++j) {
            int idx = tid + j * 128;
            int row = idx >> 2;
            int c8  = (idx & 3) << 3;
            cp_b128(&ldsA[row * GK + c8],
                    A + (size_t)(m0 + row) * Kdim + k0 + c8);
        }
        #pragma unroll
        for (int j = 0; j < 4; ++j) {
            int idx  = tid + j * 128;
            int krow = idx >> 4;
            int c4   = (idx & 15) << 2;
            v4f b4 = *(const v4f*)(W + (size_t)(k0 + krow) * N + n0 + c4);
            ldsBt[(c4 + 0) * GK + krow] = (_Float16)b4[0];
            ldsBt[(c4 + 1) * GK + krow] = (_Float16)b4[1];
            ldsBt[(c4 + 2) * GK + krow] = (_Float16)b4[2];
            ldsBt[(c4 + 3) * GK + krow] = (_Float16)b4[3];
        }
        cp_wait();
        __syncthreads();

        v16h af[2];
        #pragma unroll
        for (int mi = 0; mi < 2; ++mi)
            af[mi] = make_frag(&ldsA[(wave * 32 + mi * 16 + r) * GK + 8 * hi]);
        #pragma unroll
        for (int sub = 0; sub < 4; ++sub) {
            v16h bfrag = make_frag(&ldsBt[(sub * 16 + r) * GK + 8 * hi]);
            #pragma unroll
            for (int mi = 0; mi < 2; ++mi)
                acc[mi][sub] = __builtin_amdgcn_wmma_f32_16x16x32_f16(
                    false, af[mi], false, bfrag, (short)0, acc[mi][sub], false, false);
        }
        __syncthreads();
    }

    #pragma unroll
    for (int mi = 0; mi < 2; ++mi) {
        #pragma unroll
        for (int sub = 0; sub < 4; ++sub) {
            int col = n0 + sub * 16 + r;
            float bv = bias[col];
            #pragma unroll
            for (int v = 0; v < 8; ++v) {
                int row = m0 + wave * 32 + mi * 16 + hi * 8 + v;
                Out[(size_t)row * N + col] = acc[mi][sub][v] + bv;
            }
        }
    }
}

// ---------------- Flash attention over f16 Q/K/V -> f16 ctx ----------------
// Q pre-scaled by 1/sqrt(HD). Layouts are [B,S,H,HD] f16.
// Block: 128 threads (4 waves) covers 128 query rows of one (b,h).
// Wave covers 32 query rows (2 x 16). Key tiles of 64.
__global__ __launch_bounds__(128) void attn_kernel(
    const _Float16* __restrict__ Qw, const _Float16* __restrict__ Kw,
    const _Float16* __restrict__ Vw, _Float16* __restrict__ Ctx)
{
    __shared__ __align__(16) _Float16 ldsK[64 * 64];    // [key][hd]
    __shared__ __align__(16) _Float16 ldsVt[64 * 64];   // [hd][key]
    __shared__ __align__(16) _Float16 ldsP[4][32 * 64]; // per wave: [q][key]

    const int tid  = threadIdx.x;
    const int wave = tid >> 5;
    const int lane = tid & 31;
    const int r    = lane & 15;
    const int hi   = lane >> 4;

    const int bh = blockIdx.x;
    const int b  = bh / Hc;
    const int h  = bh % Hc;
    const int q0 = blockIdx.y * 128;

    const size_t base   = ((size_t)b * Sc * Hc + h) * HDc;  // (b, s=0, h, 0)
    const size_t rowstr = (size_t)Hc * HDc;                 // per-s stride (halves)

    // Q fragments: 2 row-halves x 2 k-steps over HD=64
    v16h aq[2][2];
    #pragma unroll
    for (int mi = 0; mi < 2; ++mi) {
        const _Float16* qp = Qw + base
            + (size_t)(q0 + wave * 32 + mi * 16 + r) * rowstr;
        #pragma unroll
        for (int ks = 0; ks < 2; ++ks)
            aq[mi][ks] = make_frag(qp + ks * 32 + 8 * hi);
    }

    float mrow[2][8], lrow[2][8];
    v8f O[2][4] = {};
    #pragma unroll
    for (int mi = 0; mi < 2; ++mi)
        #pragma unroll
        for (int v = 0; v < 8; ++v) { mrow[mi][v] = -1e30f; lrow[mi][v] = 0.0f; }

    _Float16* pw = &ldsP[wave][0];

    for (int kt0 = 0; kt0 < Sc; kt0 += 64) {
        // stage K (async straight copy) and V (transposed, through regs)
        #pragma unroll
        for (int j = 0; j < 4; ++j) {
            int idx = tid + j * 128;          // 512 chunks of 8 halves
            int key = idx >> 3;
            int c8  = (idx & 7) << 3;
            size_t g = base + (size_t)(kt0 + key) * rowstr + c8;
            cp_b128(&ldsK[key * 64 + c8], Kw + g);
            v8h vv = *(const v8h*)(Vw + g);
            #pragma unroll
            for (int t = 0; t < 8; ++t)
                ldsVt[(c8 + t) * 64 + key] = vv[t];
        }
        cp_wait();
        __syncthreads();

        // scores: S[32q x 64k] = Q(32x64) * K^T ; B-fragments reused across mi
        v8f sacc[2][4] = {};
        #pragma unroll
        for (int ks = 0; ks < 2; ++ks) {
            #pragma unroll
            for (int sub = 0; sub < 4; ++sub) {
                v16h bfrag = make_frag(&ldsK[(sub * 16 + r) * 64 + ks * 32 + 8 * hi]);
                #pragma unroll
                for (int mi = 0; mi < 2; ++mi)
                    sacc[mi][sub] = __builtin_amdgcn_wmma_f32_16x16x32_f16(
                        false, aq[mi][ks], false, bfrag, (short)0,
                        sacc[mi][sub], false, false);
            }
        }

        // online softmax per row-half; rows m = 8*hi + v live in one 16-lane half
        #pragma unroll
        for (int mi = 0; mi < 2; ++mi) {
            float scal[8];
            #pragma unroll
            for (int v = 0; v < 8; ++v) {
                float rm = fmaxf(fmaxf(sacc[mi][0][v], sacc[mi][1][v]),
                                 fmaxf(sacc[mi][2][v], sacc[mi][3][v]));
                #pragma unroll
                for (int off = 1; off < 16; off <<= 1)
                    rm = fmaxf(rm, __shfl_xor(rm, off, 32));
                float nm = fmaxf(mrow[mi][v], rm);
                scal[v] = __expf(mrow[mi][v] - nm);
                mrow[mi][v] = nm;
            }
            float rs[8];
            #pragma unroll
            for (int v = 0; v < 8; ++v) rs[v] = 0.0f;
            #pragma unroll
            for (int sub = 0; sub < 4; ++sub) {
                #pragma unroll
                for (int v = 0; v < 8; ++v) {
                    float p = __expf(sacc[mi][sub][v] - mrow[mi][v]);
                    sacc[mi][sub][v] = p;
                    rs[v] += p;
                }
            }
            #pragma unroll
            for (int v = 0; v < 8; ++v) {
                #pragma unroll
                for (int off = 1; off < 16; off <<= 1)
                    rs[v] += __shfl_xor(rs[v], off, 32);
                lrow[mi][v] = lrow[mi][v] * scal[v] + rs[v];
            }
            #pragma unroll
            for (int sub = 0; sub < 4; ++sub)
                #pragma unroll
                for (int v = 0; v < 8; ++v)
                    O[mi][sub][v] *= scal[v];
            // P: D-layout -> A-layout via per-wave LDS bounce (DS in-order/wave)
            #pragma unroll
            for (int sub = 0; sub < 4; ++sub)
                #pragma unroll
                for (int v = 0; v < 8; ++v)
                    pw[(mi * 16 + hi * 8 + v) * 64 + sub * 16 + r] =
                        (_Float16)sacc[mi][sub][v];
        }

        // O += P(32x64) * V(64x64); V B-fragments reused across mi
        #pragma unroll
        for (int ks = 0; ks < 2; ++ks) {
            v16h ap[2];
            #pragma unroll
            for (int mi = 0; mi < 2; ++mi)
                ap[mi] = make_frag(pw + (mi * 16 + r) * 64 + ks * 32 + 8 * hi);
            #pragma unroll
            for (int sub = 0; sub < 4; ++sub) {
                v16h bfrag = make_frag(&ldsVt[(sub * 16 + r) * 64 + ks * 32 + 8 * hi]);
                #pragma unroll
                for (int mi = 0; mi < 2; ++mi)
                    O[mi][sub] = __builtin_amdgcn_wmma_f32_16x16x32_f16(
                        false, ap[mi], false, bfrag, (short)0, O[mi][sub], false, false);
            }
        }
        __syncthreads();  // protect ldsK/ldsVt for next tile
    }

    // normalize and store ctx (f16, [B,S,H,HD])
    #pragma unroll
    for (int mi = 0; mi < 2; ++mi) {
        #pragma unroll
        for (int sub = 0; sub < 4; ++sub) {
            #pragma unroll
            for (int v = 0; v < 8; ++v) {
                int row = q0 + wave * 32 + mi * 16 + hi * 8 + v;
                float val = O[mi][sub][v] / lrow[mi][v];
                Ctx[base + (size_t)row * rowstr + sub * 16 + r] = (_Float16)val;
            }
        }
    }
}

extern "C" void kernel_launch(void* const* d_in, const int* in_sizes, int n_in,
                              void* d_out, int out_size, void* d_ws, size_t ws_size,
                              hipStream_t stream) {
    (void)in_sizes; (void)n_in; (void)out_size; (void)ws_size;

    const float* Xq  = (const float*)d_in[0];
    const float* Xkv = (const float*)d_in[1];
    const float* Wq  = (const float*)d_in[2];
    const float* bq  = (const float*)d_in[3];
    const float* Wk  = (const float*)d_in[4];
    const float* bk  = (const float*)d_in[5];
    const float* Wv  = (const float*)d_in[6];
    const float* bv  = (const float*)d_in[7];
    const float* Wo  = (const float*)d_in[8];
    const float* bo  = (const float*)d_in[9];
    float* out = (float*)d_out;

    const size_t elems = (size_t)BSM * Dm;  // 4096*1024
    _Float16* Qh = (_Float16*)d_ws;
    _Float16* Kh = Qh + elems;
    _Float16* Vh = Kh + elems;
    _Float16* Ch = Vh + elems;

    dim3 blk(128);
    dim3 gg(Dm / GN, BSM / GM);       // (16, 32)

    const float qscale = 0.125f;      // 1/sqrt(HD=64)
    gemm_proj_f16out<<<gg, blk, 0, stream>>>(Xq,  Wq, bq, Qh, qscale, BSM, Dm, Dm);
    gemm_proj_f16out<<<gg, blk, 0, stream>>>(Xkv, Wk, bk, Kh, 1.0f,   BSM, Dm, Dm);
    gemm_proj_f16out<<<gg, blk, 0, stream>>>(Xkv, Wv, bv, Vh, 1.0f,   BSM, Dm, Dm);

    dim3 ga(Bc * Hc, Sc / 128);       // (32, 16)
    attn_kernel<<<ga, blk, 0, stream>>>(Qh, Kh, Vh, Ch);

    gemm_out_f32<<<gg, blk, 0, stream>>>(Ch, Wo, bo, out, BSM, Dm, Dm);
}
